// MIPNetwork_80719615361737
// MI455X (gfx1250) — compile-verified, matrix operand order checked by reading, not run on hardware
//
#include <hip/hip_runtime.h>
#include <hip/hip_bf16.h>
#include <stdint.h>

// MIPNetwork forward for MI455X (gfx1250, wave32).
// Pipeline is HBM-bound (GEMM AI ~32 flop/byte << ratio needed vs 23.3 TB/s),
// so: activations stored bf16 (halves dominant traffic + SpMM gather), fused
// 2-layer MLPs on v_wmma_f32_16x16x32_bf16 with f32 accumulation (hidden tile
// never leaves LDS), SpMM accumulates in f32 via global atomics.
// Workspace: ~515 MB carved from d_ws.

#define FM    64
#define BITS  16
#define STEPS 3
#define NVAR  500000
#define NCON  250000
#define NNZE  2000000

typedef __bf16 bf16x16 __attribute__((ext_vector_type(16)));
typedef float  f32x8   __attribute__((ext_vector_type(8)));

__device__ __forceinline__ unsigned short f2bf(float f) {
  union { __bf16 b; unsigned short u; } c; c.b = (__bf16)f; return c.u;
}
__device__ __forceinline__ float bf2f(unsigned short h) {
  union { unsigned u; float f; } a; a.u = ((unsigned)h) << 16; return a.f;  // exact
}

// 16-element bf16 fragment (A or B), K-contiguous per the 16-bit operand
// layout (ISA 7.12.2): elements 0..7 at p, 8..15 at p+16 halfwords.
// Two 16-byte loads (global_load_b128 / ds_load_b128).
__device__ __forceinline__ bf16x16 frag_bf16(const unsigned short* p) {
  union { uint4 q; __bf16 h[8]; } lo, hi;
  lo.q = *(const uint4*)(p);
  hi.q = *(const uint4*)(p + 16);
  bf16x16 r;
#pragma unroll
  for (int i = 0; i < 8; ++i) { r[i] = lo.h[i]; r[i + 8] = hi.h[i]; }
  return r;
}

// Fused 2-layer MLP: out = relu(x@W1+b1) @ W2 + b2, rows a multiple of 16.
// Inputs x0..x2 are bf16 [rows][64] (concat via NIN pointers). Weights fp32 in
// global, staged transposed as bf16 into LDS.
// MODE 0: store bf16. MODE 1: scale row by extra[row] (obj_mult), store bf16.
// MODE 2: output head (NOUT=16): sigmoid(.+noise)->outf (f32), powers-dot->out2.
template <int NIN, int NOUT, int MODE>
__global__ __launch_bounds__(128) void mlp2_wmma(
    const unsigned short* x0, const unsigned short* x1, const unsigned short* x2,
    const float* w1, const float* b1, const float* w2, const float* b2,
    unsigned short* outb, float* outf, int nrows,
    const float* extra, float* out2)
{
  constexpr int K1  = NIN * FM;
  constexpr int NT2 = NOUT / 16;
  __shared__ unsigned short w1t[FM * K1];     // W1^T: [n][k], bf16
  __shared__ unsigned short w2t[NOUT * FM];   // W2^T: [n][k], bf16
  __shared__ float b1s[FM];
  __shared__ float b2s[NOUT];
  __shared__ unsigned short hid[4][16 * FM];  // per-wave hidden tile, bf16

  const int tid = threadIdx.x;
  for (int i = tid; i < K1 * FM; i += 128) {   // w1 global layout: [K1][FM]
    int k = i / FM, n = i % FM;
    w1t[n * K1 + k] = f2bf(w1[i]);
  }
  for (int i = tid; i < FM * NOUT; i += 128) { // w2 global layout: [FM][NOUT]
    int k = i / NOUT, n = i % NOUT;
    w2t[n * FM + k] = f2bf(w2[i]);
  }
  if (tid < FM)   b1s[tid] = b1[tid];
  if (tid < NOUT) b2s[tid] = b2[tid];
  __syncthreads();

  const int wave = tid >> 5;
  const int lane = tid & 31;
  const int m    = lane & 15;       // A row / B col / C col within tile
  const int hl   = lane >> 4;       // lane half: K sub-block & C row half
  const int kh8  = hl << 3;

  const int tileRow = (blockIdx.x * 4 + wave) * 16;
  const bool active = tileRow < nrows;            // nrows % 16 == 0
  const int arow    = active ? (tileRow + m) : m; // clamp loads for idle waves
  const unsigned short* xs[3] = { x0, x1, x2 };

  // Speculative prefetch of the next block's rows (global_prefetch_b8).
#pragma unroll
  for (int s = 0; s < NIN; ++s)
    __builtin_prefetch(xs[s] + ((size_t)arow + 64) * FM, 0, 0);

  f32x8 vzero = {};
  f32x8 acc[4];
#pragma unroll
  for (int i = 0; i < 4; ++i) acc[i] = vzero;

#pragma unroll
  for (int kc = 0; kc < NIN * 2; ++kc) {
    const int kbase = kc * 32;
    const unsigned short* xp =
        xs[kbase >> 6] + (size_t)arow * FM + (kbase & 63) + kh8;
    bf16x16 a = frag_bf16(xp);   // raw b128 loads, no conversion
#pragma unroll
    for (int nt = 0; nt < 4; ++nt) {
      bf16x16 b = frag_bf16(w1t + (nt * 16 + m) * K1 + kbase + kh8);
      acc[nt] = __builtin_amdgcn_wmma_f32_16x16x32_bf16(
          false, a, false, b, (short)0, acc[nt], false, false);
    }
  }

  // bias + relu -> bf16 hidden tile in LDS (C layout: lane holds col m+16nt,
  // rows r + 8*hl).
  unsigned short* h = hid[wave];
#pragma unroll
  for (int nt = 0; nt < 4; ++nt) {
    int col = nt * 16 + m;
    float bb = b1s[col];
#pragma unroll
    for (int r = 0; r < 8; ++r) {
      float v = acc[nt][r] + bb;
      v = v > 0.f ? v : 0.f;
      h[(r + hl * 8) * FM + col] = f2bf(v);
    }
  }
  __syncthreads();

  f32x8 acc2[NT2];
#pragma unroll
  for (int i = 0; i < NT2; ++i) acc2[i] = vzero;
#pragma unroll
  for (int kc = 0; kc < 2; ++kc) {
    bf16x16 a = frag_bf16(h + m * FM + kc * 32 + kh8);
#pragma unroll
    for (int nt = 0; nt < NT2; ++nt) {
      bf16x16 b = frag_bf16(w2t + (nt * 16 + m) * FM + kc * 32 + kh8);
      acc2[nt] = __builtin_amdgcn_wmma_f32_16x16x32_bf16(
          false, a, false, b, (short)0, acc2[nt], false, false);
    }
  }

  if (!active) return;

  if (MODE == 2) {
    const int col = m;
    const float pw = (float)(1u << col);   // powers = 2^bit
#pragma unroll
    for (int r = 0; r < 8; ++r) {
      int orow = tileRow + r + hl * 8;
      float v = acc2[0][r] + b2s[col] + extra[(size_t)orow * BITS + col];
      float s = 1.0f / (1.0f + __expf(-v));
      outf[(size_t)orow * BITS + col] = s;
      float d = s * pw;
#pragma unroll
      for (int off = 1; off < 16; off <<= 1) d += __shfl_xor(d, off, 32);
      if (m == 0) out2[orow] = d;
    }
  } else {
#pragma unroll
    for (int nt = 0; nt < NT2; ++nt) {
      int col = nt * 16 + m;
      float bb = b2s[col];
#pragma unroll
      for (int r = 0; r < 8; ++r) {
        int orow = tileRow + r + hl * 8;
        float v = acc2[nt][r] + bb;
        if (MODE == 1) v *= extra[orow];
        outb[(size_t)orow * NOUT + col] = f2bf(v);
      }
    }
  }
}

__global__ void fill_f32(float* p, float v, long long n) {
  long long i = (long long)blockIdx.x * blockDim.x + threadIdx.x;
  long long st = (long long)gridDim.x * blockDim.x;
  for (; i < n; i += st) p[i] = v;
}

__global__ void fill_u16(unsigned short* p, unsigned short v, long long n) {
  long long i = (long long)blockIdx.x * blockDim.x + threadIdx.x;
  long long st = (long long)gridDim.x * blockDim.x;
  for (; i < n; i += st) p[i] = v;
}

__global__ void f32_to_bf16_k(const float* src, unsigned short* dst, long long n) {
  long long i = (long long)blockIdx.x * blockDim.x + threadIdx.x;
  long long st = (long long)gridDim.x * blockDim.x;
  for (; i < n; i += st) dst[i] = f2bf(src[i]);
}

__global__ void scatter_add_1d(const int* idx, const float* vals, float* out, int n) {
  int i = blockIdx.x * blockDim.x + threadIdx.x;
  if (i < n) atomicAdd(&out[idx[i]], vals[i]);
}

// dst[sidx[e], :] += vals[e] * src[gidx[e], :]  (thread per edge-feature,
// bf16 gather, f32 atomic accumulate)
__global__ void spmm_scatter(const int* gidx, const int* sidx, const float* vals,
                             const unsigned short* src, float* dst, int nnz) {
  long long t = (long long)blockIdx.x * blockDim.x + threadIdx.x;
  if (t >= (long long)nnz * FM) return;
  int e = (int)(t >> 6);
  int col = (int)(t & 63);
  float v = vals[e];
  int g = gidx[e], s = sidx[e];
  atomicAdd(&dst[(size_t)s * FM + col], v * bf2f(src[(size_t)g * FM + col]));
}

// closs = relu(v2c - cv) as bf16
__global__ void const_loss_k(const float* v2c, const float* cv,
                             unsigned short* closs, int c) {
  long long t = (long long)blockIdx.x * blockDim.x + threadIdx.x;
  if (t >= (long long)c * FM) return;
  int row = (int)(t >> 6);
  float v = v2c[t] - cv[row];
  closs[t] = f2bf(v > 0.f ? v : 0.f);
}

__global__ void colsum_k(const unsigned short* x, int nrows, float* colsum) {
  int col = threadIdx.x & 63;
  int sub = threadIdx.x >> 6;
  float s = 0.f;
  for (int r = blockIdx.x * 4 + sub; r < nrows; r += gridDim.x * 4)
    s += bf2f(x[(size_t)r * FM + col]);
  atomicAdd(&colsum[col], s);
}

__global__ void center_sumsq_k(unsigned short* x, int nrows,
                               const float* colsum, float* sumsq) {
  __shared__ float mean[FM];
  __shared__ float red[256];
  if (threadIdx.x < FM) mean[threadIdx.x] = colsum[threadIdx.x] / (float)nrows;
  __syncthreads();
  int col = threadIdx.x & 63;
  int sub = threadIdx.x >> 6;
  float local = 0.f;
  for (int r = blockIdx.x * 4 + sub; r < nrows; r += gridDim.x * 4) {
    float v = bf2f(x[(size_t)r * FM + col]) - mean[col];
    x[(size_t)r * FM + col] = f2bf(v);
    local += v * v;
  }
  red[threadIdx.x] = local;
  __syncthreads();
  for (int s = 128; s > 0; s >>= 1) {
    if (threadIdx.x < s) red[threadIdx.x] += red[threadIdx.x + s];
    __syncthreads();
  }
  if (threadIdx.x == 0) atomicAdd(sumsq, red[0]);
}

__global__ void scale_k(unsigned short* x, long long n, const float* sumsq,
                        float inv_rows) {
  float scale = rsqrtf(1e-6f + sumsq[0] * inv_rows);
  long long i = (long long)blockIdx.x * blockDim.x + threadIdx.x;
  long long st = (long long)gridDim.x * blockDim.x;
  for (; i < n; i += st) x[i] = f2bf(bf2f(x[i]) * scale);
}

extern "C" void kernel_launch(void* const* d_in, const int* in_sizes, int n_in,
                              void* d_out, int out_size, void* d_ws, size_t ws_size,
                              hipStream_t stream) {
  // params flattened in dict insertion order (w,b pairs):
  const float* cu1_w = (const float*)d_in[0];
  const float* cu1_b = (const float*)d_in[1];
  const float* cu2_w = (const float*)d_in[2];
  const float* cu2_b = (const float*)d_in[3];
  const float* qc1_w = (const float*)d_in[4];
  const float* qc1_b = (const float*)d_in[5];
  const float* qc2_w = (const float*)d_in[6];
  const float* qc2_b = (const float*)d_in[7];
  const float* qo1_w = (const float*)d_in[8];
  const float* qo1_b = (const float*)d_in[9];
  const float* qo2_w = (const float*)d_in[10];
  const float* qo2_b = (const float*)d_in[11];
  const float* vu1_w = (const float*)d_in[12];
  const float* vu1_b = (const float*)d_in[13];
  const float* vu2_w = (const float*)d_in[14];
  const float* vu2_b = (const float*)d_in[15];
  const float* o1_w  = (const float*)d_in[16];
  const float* o1_b  = (const float*)d_in[17];
  const float* o2_w  = (const float*)d_in[18];
  const float* o2_b  = (const float*)d_in[19];
  const int*   edge_rows = (const int*)d_in[20];
  const int*   edge_cols = (const int*)d_in[21];
  const float* edge_vals = (const float*)d_in[22];
  const float* const_values = (const float*)d_in[23];
  const int*   obj_rows = (const int*)d_in[24];
  const float* obj_vals = (const float*)d_in[25];
  const float* noise    = (const float*)d_in[26];

  float* binary  = (float*)d_out;                          // [STEPS][V][BITS]
  float* decimal = binary + (size_t)STEPS * NVAR * BITS;   // [STEPS][V]

  // Workspace carve-out (256B aligned), total ~515 MB.
  char* wsp = (char*)d_ws;
  auto carve = [&](size_t bytes) -> void* {
    char* p = wsp; wsp += (bytes + 255) & ~(size_t)255; return p;
  };
  unsigned short* vars   = (unsigned short*)carve((size_t)NVAR * FM * 2);
  unsigned short* cons   = (unsigned short*)carve((size_t)NCON * FM * 2);
  unsigned short* cquery = (unsigned short*)carve((size_t)NVAR * FM * 2);
  unsigned short* closs  = (unsigned short*)carve((size_t)NCON * FM * 2);
  unsigned short* oquery = (unsigned short*)carve((size_t)NVAR * FM * 2);
  unsigned short* c2vb   = (unsigned short*)carve((size_t)NVAR * FM * 2);
  float* v2c   = (float*)carve((size_t)NCON * FM * 4);   // f32 SpMM accum
  float* c2v   = (float*)carve((size_t)NVAR * FM * 4);   // f32 SpMM accum
  float* objm  = (float*)carve((size_t)NVAR * 4);
  float* stats = (float*)carve(65 * 4);                  // colsum[64] + sumsq

  const int gV = (NVAR / 16 + 3) / 4;   // 16-row wave tiles, 4 waves/block
  const int gC = (NCON / 16 + 3) / 4;
  const int FB = 2048;                  // grid-stride blocks

  // init: variables = constraints = 1 (bf16 1.0 = 0x3F80),
  // obj_mult = segment_sum(obj_vals, obj_rows)
  fill_u16<<<FB, 256, 0, stream>>>(vars, (unsigned short)0x3F80, (long long)NVAR * FM);
  fill_u16<<<FB, 256, 0, stream>>>(cons, (unsigned short)0x3F80, (long long)NCON * FM);
  fill_f32<<<FB, 256, 0, stream>>>(objm, 0.0f, (long long)NVAR);
  scatter_add_1d<<<(NVAR + 255) / 256, 256, 0, stream>>>(obj_rows, obj_vals, objm, NVAR);

  auto pairnorm = [&](unsigned short* x, int nrows) {
    fill_f32<<<1, 65, 0, stream>>>(stats, 0.0f, 65);
    colsum_k<<<512, 256, 0, stream>>>(x, nrows, stats);
    center_sumsq_k<<<512, 256, 0, stream>>>(x, nrows, stats, stats + 64);
    scale_k<<<FB, 256, 0, stream>>>(x, (long long)nrows * FM, stats + 64,
                                    1.0f / nrows);
  };

  const int spmmBlocks = (int)(((long long)NNZE * FM + 255) / 256);

  for (int step = 0; step < STEPS; ++step) {
    // const_query = mlp2(vars; qc)
    mlp2_wmma<1, 64, 0><<<gV, 128, 0, stream>>>(
        vars, nullptr, nullptr, qc1_w, qc1_b, qc2_w, qc2_b,
        cquery, nullptr, NVAR, nullptr, nullptr);
    // v2c = segsum(edge_vals * cquery[edge_rows] -> edge_cols)
    fill_f32<<<FB, 256, 0, stream>>>(v2c, 0.0f, (long long)NCON * FM);
    spmm_scatter<<<spmmBlocks, 256, 0, stream>>>(edge_rows, edge_cols, edge_vals,
                                                 cquery, v2c, NNZE);
    // const_loss = relu(v2c - cv) -> bf16
    const_loss_k<<<(int)(((long long)NCON * FM + 255) / 256), 256, 0, stream>>>(
        v2c, const_values, closs, NCON);
    // constraints = pairnorm(mlp2([cons, closs]; cu))   (in-place, per-row)
    mlp2_wmma<2, 64, 0><<<gC, 128, 0, stream>>>(
        cons, closs, nullptr, cu1_w, cu1_b, cu2_w, cu2_b,
        cons, nullptr, NCON, nullptr, nullptr);
    pairnorm(cons, NCON);
    // obj_query = mlp2(vars; qo) * obj_mult  (pre-update vars)
    mlp2_wmma<1, 64, 1><<<gV, 128, 0, stream>>>(
        vars, nullptr, nullptr, qo1_w, qo1_b, qo2_w, qo2_b,
        oquery, nullptr, NVAR, objm, nullptr);
    // c2v = segsum(edge_vals * cons[edge_cols] -> edge_rows)   (updated cons)
    fill_f32<<<FB, 256, 0, stream>>>(c2v, 0.0f, (long long)NVAR * FM);
    spmm_scatter<<<spmmBlocks, 256, 0, stream>>>(edge_cols, edge_rows, edge_vals,
                                                 cons, c2v, NNZE);
    f32_to_bf16_k<<<FB, 256, 0, stream>>>(c2v, c2vb, (long long)NVAR * FM);
    // variables = pairnorm(mlp2([vars, c2vb, oquery]; vu))   (in-place)
    mlp2_wmma<3, 64, 0><<<gV, 128, 0, stream>>>(
        vars, c2vb, oquery, vu1_w, vu1_b, vu2_w, vu2_b,
        vars, nullptr, NVAR, nullptr, nullptr);
    pairnorm(vars, NVAR);
    // output head: binary = sigmoid(mlp2(vars; o) + noise), decimal = powers.binary
    mlp2_wmma<1, 16, 2><<<gV, 128, 0, stream>>>(
        vars, nullptr, nullptr, o1_w, o1_b, o2_w, o2_b,
        nullptr, binary + (size_t)step * NVAR * BITS, NVAR,
        noise + (size_t)step * NVAR * BITS,
        decimal + (size_t)step * NVAR);
    // EMA with stop_gradient is identity in the forward pass -> skipped.
  }
}
